// MVAE_77721728188872
// MI455X (gfx1250) — matrix operand order repaired and checked by reading, not compiled, not run on hardware
//
#include <hip/hip_runtime.h>
#include <hip/hip_bf16.h>

#define LDIM 6
#define HDIM 64
#define VDIM 51
#define KNB  64
#define EPSF 1e-8f

#define WAVES_PER_BLOCK 8
#define ROWS_PER_WAVE   16
#define ROWS_PER_BLOCK  (WAVES_PER_BLOCK * ROWS_PER_WAVE)

typedef __attribute__((ext_vector_type(16))) _Float16 v16h;
typedef __attribute__((ext_vector_type(8)))  _Float16 v8h;
typedef __attribute__((ext_vector_type(8)))  float    v8f;

// 16-bit A/B k order used by this code (not the ISA table order): lane = (M | g<<4), slot h -> K index.
// Slots 0..15 <-> K = 16g + h                (contiguous 16-half run)
__device__ __forceinline__ int a_kidx(int h, int g) {
    return 16 * g + h;
}

// Fast reciprocal: single v_rcp_f32 instead of IEEE div expansion.
__device__ __forceinline__ float frcp(float x) {
    return __builtin_amdgcn_rcpf(x);
}

__device__ __forceinline__ float swishf(float x) {
    return x * frcp(1.0f + __expf(-x));
}

__device__ __forceinline__ v8f wmma_f16(v16h a, v16h b, v8f c) {
    // D = A(16x32 f16) * B(32x16 f16) + C(16x16 f32)
    return __builtin_amdgcn_wmma_f32_16x16x32_f16(
        /*neg_a=*/false, a, /*neg_b=*/false, b,
        /*c_mod=*/(short)0, c, /*reuse_a=*/false, /*reuse_b=*/false);
}

// Load a WMMA A-operand (16 halves) from a row-major [16][64] f16 LDS tile.
// base points at row M; khalf = 0 for K=0..31 operand, 32 for K=32..63.
__device__ __forceinline__ v16h load_a_frag(const _Float16* base, int g, int khalf) {
    v8h p0 = *(const v8h*)(base + khalf + 16 * g);
    v8h p1 = *(const v8h*)(base + khalf + 16 * g + 8);
    v16h a;
    #pragma unroll
    for (int j = 0; j < 8; ++j) { a[j] = p0[j]; a[j + 8] = p1[j]; }
    return a;
}

// ---------------------------------------------------------------------------
// Kernel A: encoder. obs(N,2) -> h64 -> h64 -> (mu, logvar) each (N,12) f32.
// ---------------------------------------------------------------------------
__global__ __launch_bounds__(256) void mvae_encoder(
    const float* __restrict__ obs,
    const float* __restrict__ We1, const float* __restrict__ be1,
    const float* __restrict__ We2, const float* __restrict__ be2,
    const float* __restrict__ We3, const float* __restrict__ be3,
    float* __restrict__ mu, float* __restrict__ logvar, int N)
{
    // Weight tiles stored TRANSPOSED: sW2t[nc*64 + k], so a B-fragment
    // (fixed nc, K=16g..16g+15) is one contiguous, 32B-aligned v16h load.
    __shared__ __align__(32) _Float16 sW2t[HDIM * HDIM];   // 8 KB
    __shared__ __align__(32) _Float16 sW3t[32 * HDIM];     // 4 KB, nc>=24 zero-padded
    __shared__ float sW1[2 * HDIM];
    __shared__ float sb1[HDIM], sb2[HDIM], sb3[24];
    __shared__ __align__(32) _Float16 sHt[WAVES_PER_BLOCK][ROWS_PER_WAVE * HDIM]; // 16 KB

    const int tid = threadIdx.x;
    // Coalesced global read, scattered (one-time) LDS write for the transpose.
    for (int i = tid; i < HDIM * HDIM; i += 256) {
        int k = i >> 6, nc = i & 63;
        sW2t[nc * HDIM + k] = (_Float16)We2[i];
    }
    for (int i = tid; i < 32 * HDIM; i += 256) {
        int nc = i >> 6, k = i & 63;
        sW3t[i] = (nc < 24) ? (_Float16)We3[k * 24 + nc] : (_Float16)0.0f;
    }
    for (int i = tid; i < 2 * HDIM; i += 256) sW1[i] = We1[i];
    for (int i = tid; i < HDIM;     i += 256) { sb1[i] = be1[i]; sb2[i] = be2[i]; }
    for (int i = tid; i < 24;       i += 256) sb3[i] = be3[i];
    __syncthreads();

    const int wid    = tid >> 5;
    const int lane   = tid & 31;
    const int M      = lane & 15;
    const int g      = lane >> 4;
    const int ncol15 = lane & 15;
    const int rowBase = (blockIdx.x * WAVES_PER_BLOCK + wid) * ROWS_PER_WAVE;

    int r = rowBase + M; if (r >= N) r = N - 1;   // clamp: keep EXEC all-ones for WMMA
    const float o0 = obs[(size_t)r * 2 + 0];
    const float o1 = obs[(size_t)r * 2 + 1];

    // Layer 1 (K=2): compute swish(obs@We1+be1) directly in A-fragment layout.
    v16h aLo, aHi;
    #pragma unroll
    for (int h = 0; h < 16; ++h) {
        int k = a_kidx(h, g);
        float v0 = o0 * sW1[k]        + o1 * sW1[HDIM + k]        + sb1[k];
        float v1 = o0 * sW1[k + 32]   + o1 * sW1[HDIM + k + 32]   + sb1[k + 32];
        aLo[h] = (_Float16)swishf(v0);
        aHi[h] = (_Float16)swishf(v1);
    }

    // Layer 2 (64x64): 4 N-tiles, each K=64 as two 16x16x32 WMMAs.
    #pragma unroll
    for (int nt = 0; nt < 4; ++nt) {
        const int nc = nt * 16 + ncol15;
        const v16h bLo = *(const v16h*)&sW2t[nc * HDIM + 16 * g];       // K=16g..16g+15
        const v16h bHi = *(const v16h*)&sW2t[nc * HDIM + 32 + 16 * g];  // +32
        v8f c;
        const float bias = sb2[nc];
        #pragma unroll
        for (int v = 0; v < 8; ++v) c[v] = bias;
        c = wmma_f16(aLo, bLo, c);
        c = wmma_f16(aHi, bHi, c);
        #pragma unroll
        for (int v = 0; v < 8; ++v) {            // D: row = v + g*8, col = nc
            int row = v + g * 8;
            sHt[wid][row * HDIM + nc] = (_Float16)swishf(c[v]);
        }
    }
    __syncthreads();

    // Re-gather h2 in A-fragment layout: 4x ds_load_b128.
    const _Float16* hrow = &sHt[wid][M * HDIM];
    const v16h a2Lo = load_a_frag(hrow, g, 0);
    const v16h a2Hi = load_a_frag(hrow, g, 32);

    // Layer 3 (64x24): 2 N-tiles, second half zero-padded weights, masked stores.
    #pragma unroll
    for (int nt = 0; nt < 2; ++nt) {
        const int nc = nt * 16 + ncol15;
        const bool valid = nc < 24;
        const v16h bLo = *(const v16h*)&sW3t[nc * HDIM + 16 * g];
        const v16h bHi = *(const v16h*)&sW3t[nc * HDIM + 32 + 16 * g];
        v8f c;
        const float bias = valid ? sb3[nc] : 0.0f;
        #pragma unroll
        for (int v = 0; v < 8; ++v) c[v] = bias;
        c = wmma_f16(a2Lo, bLo, c);
        c = wmma_f16(a2Hi, bHi, c);
        #pragma unroll
        for (int v = 0; v < 8; ++v) {
            int row = v + g * 8;
            int rr  = rowBase + row;
            if (rr < N && valid) {
                if (nc < 12) mu[(size_t)rr * 12 + nc]            = c[v];
                else         logvar[(size_t)rr * 12 + (nc - 12)] = c[v];
            }
        }
    }
}

// ---------------------------------------------------------------------------
// Kernel B: PoE over 66 experts (gathered through L2-resident mu/lv table),
// branch select, reparameterize. Writes z to ws, mus/logvars to d_out.
// Skip rows avoid all neighbor-index traffic entirely.
// ---------------------------------------------------------------------------
__global__ __launch_bounds__(256) void mvae_poe(
    const float* __restrict__ mu, const float* __restrict__ lv,
    const float* __restrict__ p,
    const float* __restrict__ eps_skip, const float* __restrict__ eps_poe,
    const int* __restrict__ nsched, const int* __restrict__ nobs,
    float* __restrict__ zbuf, float* __restrict__ out_mus,
    float* __restrict__ out_lvs, int N)
{
    const int idx = blockIdx.x * blockDim.x + threadIdx.x;
    if (idx >= 2 * N) return;
    const int n = idx >> 1, m = idx & 1;
    const size_t row = (size_t)idx;               // row == n*2+m in (N,2,*) layouts

    float mself[LDIM], lself[LDIM], msel[LDIM], lsel[LDIM];
    {   // 24B rows are float2-aligned: 3x b64 loads each.
        const float2* mr = (const float2*)(mu + row * LDIM);
        const float2* lr = (const float2*)(lv + row * LDIM);
        #pragma unroll
        for (int q = 0; q < 3; ++q) {
            float2 a = mr[q], b = lr[q];
            mself[2*q] = a.x; mself[2*q+1] = a.y;
            lself[2*q] = b.x; lself[2*q+1] = b.y;
        }
    }

    const bool skip = p[n] < 0.5f;
    if (skip) {
        #pragma unroll
        for (int l = 0; l < LDIM; ++l) { msel[l] = mself[l]; lsel[l] = lself[l]; }
    } else {
        float sT[LDIM], mT[LDIM];
        const float Tpri = frcp(1.0f + EPSF);     // prior expert: mu=0, logvar=0
        #pragma unroll
        for (int l = 0; l < LDIM; ++l) {
            float T = frcp(__expf(lself[l]) + EPSF);
            sT[l] = Tpri + T;
            mT[l] = mself[l] * T;
        }
        const int* sp = nsched + row * KNB;
        const int* op = nobs   + row * KNB;
        for (int k = 0; k < KNB; ++k) {
            const size_t r2 = (size_t)sp[k] * 2 + (size_t)op[k];
            const float2* nm = (const float2*)(mu + r2 * LDIM);
            const float2* nl = (const float2*)(lv + r2 * LDIM);
            #pragma unroll
            for (int q = 0; q < 3; ++q) {
                float2 mv = nm[q], lvv = nl[q];
                float T0 = frcp(__expf(lvv.x) + EPSF);
                float T1 = frcp(__expf(lvv.y) + EPSF);
                sT[2*q]   += T0;  mT[2*q]   += mv.x * T0;
                sT[2*q+1] += T1;  mT[2*q+1] += mv.y * T1;
            }
        }
        #pragma unroll
        for (int l = 0; l < LDIM; ++l) {
            msel[l] = mT[l] * frcp(sT[l]);
            lsel[l] = -__logf(sT[l]);
        }
    }

    const float* eps = (skip ? eps_skip : eps_poe) + row * LDIM;
    float* zr = zbuf + row * LDIM;
    float* om = out_mus + (size_t)n * 12 + m * LDIM;
    float* ol = out_lvs + (size_t)n * 12 + m * LDIM;
    #pragma unroll
    for (int l = 0; l < LDIM; ++l) {
        zr[l] = msel[l] + eps[l] * __expf(0.5f * lsel[l]);
        om[l] = msel[l];
        ol[l] = lsel[l];
    }
}

// ---------------------------------------------------------------------------
// Kernel C: decoder. z(2N,6) -> h64 -> h64 -> 51, written into recs (N,102).
// ---------------------------------------------------------------------------
__global__ __launch_bounds__(256) void mvae_decoder(
    const float* __restrict__ zbuf,
    const float* __restrict__ Wd1, const float* __restrict__ bd1,
    const float* __restrict__ Wd2, const float* __restrict__ bd2,
    const float* __restrict__ Wd3, const float* __restrict__ bd3,
    float* __restrict__ recs, int N)
{
    __shared__ __align__(32) _Float16 sW2t[HDIM * HDIM];   // 8 KB, transposed
    __shared__ __align__(32) _Float16 sW3t[HDIM * HDIM];   // 8 KB, nc>=51 zero-padded
    __shared__ float sW1[LDIM * HDIM];                     // fp32 for the K=6 VALU layer
    __shared__ float sb1[HDIM], sb2[HDIM], sb3[VDIM];
    __shared__ __align__(32) _Float16 sHt[WAVES_PER_BLOCK][ROWS_PER_WAVE * HDIM];

    const int tid = threadIdx.x;
    for (int i = tid; i < HDIM * HDIM; i += 256) {
        int k = i >> 6, nc = i & 63;
        sW2t[nc * HDIM + k] = (_Float16)Wd2[i];
    }
    for (int i = tid; i < HDIM * HDIM; i += 256) {
        int nc = i >> 6, k = i & 63;
        sW3t[i] = (nc < VDIM) ? (_Float16)Wd3[k * VDIM + nc] : (_Float16)0.0f;
    }
    for (int i = tid; i < LDIM * HDIM; i += 256) sW1[i] = Wd1[i];
    for (int i = tid; i < HDIM;        i += 256) { sb1[i] = bd1[i]; sb2[i] = bd2[i]; }
    for (int i = tid; i < VDIM;        i += 256) sb3[i] = bd3[i];
    __syncthreads();

    const int wid    = tid >> 5;
    const int lane   = tid & 31;
    const int M      = lane & 15;
    const int g      = lane >> 4;
    const int ncol15 = lane & 15;
    const int totalRows = 2 * N;
    const int rowBase = (blockIdx.x * WAVES_PER_BLOCK + wid) * ROWS_PER_WAVE;

    int r = rowBase + M; if (r >= totalRows) r = totalRows - 1;
    float zr[LDIM];
    {   // 24B-aligned row: 3x b64 loads.
        const float2* zp = (const float2*)(zbuf + (size_t)r * LDIM);
        #pragma unroll
        for (int q = 0; q < 3; ++q) { float2 t = zp[q]; zr[2*q] = t.x; zr[2*q+1] = t.y; }
    }

    // Layer d1 (K=6): direct VALU into A-fragment layout.
    v16h aLo, aHi;
    #pragma unroll
    for (int h = 0; h < 16; ++h) {
        int k = a_kidx(h, g);
        float v0 = sb1[k], v1 = sb1[k + 32];
        #pragma unroll
        for (int i = 0; i < LDIM; ++i) {
            v0 += zr[i] * sW1[i * HDIM + k];
            v1 += zr[i] * sW1[i * HDIM + k + 32];
        }
        aLo[h] = (_Float16)swishf(v0);
        aHi[h] = (_Float16)swishf(v1);
    }

    // Layer d2 (64x64).
    #pragma unroll
    for (int nt = 0; nt < 4; ++nt) {
        const int nc = nt * 16 + ncol15;
        const v16h bLo = *(const v16h*)&sW2t[nc * HDIM + 16 * g];
        const v16h bHi = *(const v16h*)&sW2t[nc * HDIM + 32 + 16 * g];
        v8f c;
        const float bias = sb2[nc];
        #pragma unroll
        for (int v = 0; v < 8; ++v) c[v] = bias;
        c = wmma_f16(aLo, bLo, c);
        c = wmma_f16(aHi, bHi, c);
        #pragma unroll
        for (int v = 0; v < 8; ++v) {
            int row = v + g * 8;
            sHt[wid][row * HDIM + nc] = (_Float16)swishf(c[v]);
        }
    }
    __syncthreads();

    const _Float16* hrow = &sHt[wid][M * HDIM];
    const v16h a2Lo = load_a_frag(hrow, g, 0);
    const v16h a2Hi = load_a_frag(hrow, g, 32);

    // Layer d3 (64x51): 4 N-tiles, zero-padded weights, masked stores.
    #pragma unroll
    for (int nt = 0; nt < 4; ++nt) {
        const int nc = nt * 16 + ncol15;
        const bool valid = nc < VDIM;
        const v16h bLo = *(const v16h*)&sW3t[nc * HDIM + 16 * g];
        const v16h bHi = *(const v16h*)&sW3t[nc * HDIM + 32 + 16 * g];
        v8f c;
        const float bias = valid ? sb3[nc] : 0.0f;
        #pragma unroll
        for (int v = 0; v < 8; ++v) c[v] = bias;
        c = wmma_f16(a2Lo, bLo, c);
        c = wmma_f16(a2Hi, bHi, c);
        #pragma unroll
        for (int v = 0; v < 8; ++v) {
            int row = v + g * 8;
            int rr  = rowBase + row;
            if (rr < totalRows && valid) {
                int n = rr >> 1, m = rr & 1;
                recs[(size_t)n * 102 + m * VDIM + nc] = c[v];
            }
        }
    }
}

// ---------------------------------------------------------------------------
extern "C" void kernel_launch(void* const* d_in, const int* in_sizes, int n_in,
                              void* d_out, int out_size, void* d_ws, size_t ws_size,
                              hipStream_t stream)
{
    const float* obs      = (const float*)d_in[0];
    const float* p        = (const float*)d_in[1];
    const float* eps_skip = (const float*)d_in[2];
    const float* eps_poe  = (const float*)d_in[3];
    const int*   nsched   = (const int*)d_in[4];
    const int*   nobs     = (const int*)d_in[5];
    const float* We1 = (const float*)d_in[6];
    const float* be1 = (const float*)d_in[7];
    const float* We2 = (const float*)d_in[8];
    const float* be2 = (const float*)d_in[9];
    const float* We3 = (const float*)d_in[10];
    const float* be3 = (const float*)d_in[11];
    const float* Wd1 = (const float*)d_in[12];
    const float* bd1 = (const float*)d_in[13];
    const float* Wd2 = (const float*)d_in[14];
    const float* bd2 = (const float*)d_in[15];
    const float* Wd3 = (const float*)d_in[16];
    const float* bd3 = (const float*)d_in[17];

    const int N = in_sizes[0] / 2;

    // Workspace layout (f32): mu (N*12) | logvar (N*12) | z (N*12)
    float* ws  = (float*)d_ws;
    float* mu  = ws;
    float* lvb = ws + (size_t)N * 12;
    float* zb  = ws + (size_t)N * 24;

    // Output layout: recs (N*102) | mus (N*12) | logvars (N*12)
    float* out      = (float*)d_out;
    float* out_recs = out;
    float* out_mus  = out + (size_t)N * 102;
    float* out_lvs  = out + (size_t)N * 114;

    const int blkA = (N + ROWS_PER_BLOCK - 1) / ROWS_PER_BLOCK;
    mvae_encoder<<<blkA, 256, 0, stream>>>(obs, We1, be1, We2, be2, We3, be3,
                                           mu, lvb, N);

    const int blkB = (2 * N + 255) / 256;
    mvae_poe<<<blkB, 256, 0, stream>>>(mu, lvb, p, eps_skip, eps_poe,
                                       nsched, nobs, zb, out_mus, out_lvs, N);

    const int blkC = (2 * N + ROWS_PER_BLOCK - 1) / ROWS_PER_BLOCK;
    mvae_decoder<<<blkC, 256, 0, stream>>>(zb, Wd1, bd1, Wd2, bd2, Wd3, bd3,
                                           out_recs, N);

    (void)n_in; (void)out_size; (void)ws_size;
}
